// AdderVDSR_8693013807227
// MI455X (gfx1250) — matrix operand order repaired and checked by reading, not compiled, not run on hardware
//
#include <hip/hip_runtime.h>

typedef __attribute__((ext_vector_type(16))) _Float16 v16h;
typedef __attribute__((ext_vector_type(8)))  float    v8f;
typedef __attribute__((ext_vector_type(4)))  unsigned int u32x4;
typedef __attribute__((ext_vector_type(8)))  int      i32x8;
typedef __attribute__((ext_vector_type(4)))  int      i32x4;

#define IMG 96
#define PW  98              // padded width/height ([C][98][98], 1-px zero ring)
#define PP  (PW * PW)       // 9604
#define TW  32
#define TH  8
#define NTX (IMG / TW)      // 3
#define NTY (IMG / TH)      // 12
#define TPC ((TH + 2) * (TW + 2))   // 340 floats per channel in LDS tile

// ---------------------------------------------------------------------------
// TDM: one tensor_load_to_lds moves the whole halo tile
//   tile_dim = {34 (x), 10 (y), C (channels)}  from padded [C][98][98] global
//   into packed LDS [C][10][34]. Descriptor per CDNA5 ISA 8.3-8.6.
// Toolchain uses the 6-arg builtin form (extra int32x8 operand before cpol).
// ---------------------------------------------------------------------------
__device__ __forceinline__
void tdm_tile_load(const float* __restrict__ gsrc, unsigned lds_byte_addr, int C)
{
    unsigned long long ga = (unsigned long long)(uintptr_t)gsrc;

    u32x4 g0;
    g0[0] = 1u;                                   // count=1, user desc, no gather
    g0[1] = lds_byte_addr;                        // lds_addr (bytes)
    g0[2] = (unsigned)(ga & 0xFFFFFFFFu);         // global_addr[31:0]
    g0[3] = (unsigned)((ga >> 32) & 0x1FFFFFFu)   // global_addr[56:32]
          | (2u << 30);                           // type = 2 ("image")

    i32x8 g1;
    g1[0] = (2 << 16);                            // wg_mask=0, data_size=2 (4B)
    g1[1] = (PW & 0xFFFF) << 16;                  // tensor_dim0[15:0]
    g1[2] = (PW >> 16) | ((PW & 0xFFFF) << 16);   // dim0[31:16] | dim1[15:0]
    g1[3] = (PW >> 16) | ((TW + 2) << 16);        // dim1[31:16] | tile_dim0=34
    g1[4] = (TH + 2) | (C << 16);                 // tile_dim1=10 | tile_dim2=C
    g1[5] = PW;                                   // tensor_dim0_stride[31:0]
    g1[6] = ((PP & 0xFFFF) << 16);                // d0_stride[47:32]=0 | d1_stride[15:0]
    g1[7] = (PP >> 16);                           // tensor_dim1_stride[47:16]

    i32x4 g2;
    g2[0] = C;                                    // tensor_dim2
    g2[1] = 1;                                    // tensor_dim3 (unused)
    g2[2] = PP;                                   // tensor_dim2_stride[31:0]
    g2[3] = 0;                                    // stride hi | tile_dim3=0

    i32x4 g3 = {0, 0, 0, 0};
    i32x8 gx = {0, 0, 0, 0, 0, 0, 0, 0};          // extra operand (unused, zero)

    __builtin_amdgcn_tensor_load_to_lds(g0, g1, g2, g3, gx, /*cpol=*/0);
}

// ---------------------------------------------------------------------------
// Zero padded workspace buffers (pads must be 0; interiors rewritten anyway)
// ---------------------------------------------------------------------------
__global__ __launch_bounds__(256)
void zero_buf(float* __restrict__ p, int n)
{
    int i = blockIdx.x * 256 + threadIdx.x;
    if (i < n) p[i] = 0.0f;
}

// ---------------------------------------------------------------------------
// Kernel 1: true conv3x3 (w_up) + fused pixel_shuffle(3) via WMMA.
//   D[o,p] = sum_K W[o,K] * im2col(x)[K,p];  M=27->2 tiles, K=27->32, N=1024.
// Writes the 96x96 residual into the padded [3][98][98] buffer.
// ---------------------------------------------------------------------------
__global__ __launch_bounds__(32)
void upconv_ps_wmma(const float* __restrict__ x,     // [3,32,32]
                    const float* __restrict__ w_up,  // [27,27]
                    float* __restrict__ residual)    // padded [3,98,98]
{
    const int lane  = threadIdx.x;
    const int mtile = blockIdx.x & 1;
    const int ntile = blockIdx.x >> 1;

    // A fragment: 16-bit A 16x32 layout (ISA 7.12.2)
    const int mrow   = lane & 15;
    const int kbaseA = (lane < 16) ? 0 : 8;
    const int orow   = mtile * 16 + mrow;
    v16h a;
#pragma unroll
    for (int h = 0; h < 16; ++h) {
        int j = h >> 1, lo = h & 1;
        int K = (j < 4) ? (kbaseA + 2 * j + lo) : (kbaseA + 16 + 2 * (j - 4) + lo);
        float v = 0.0f;
        if (orow < 27 && K < 27) v = w_up[orow * 27 + K];
        a[h] = (_Float16)v;
    }

    // B fragment: 16-bit B 32x16 layout
    const int ncol   = lane & 15;
    const int p      = ntile * 16 + ncol;
    const int py     = p >> 5, px = p & 31;
    const int kbaseB = (lane < 16) ? 0 : 8;
    v16h b;
#pragma unroll
    for (int h = 0; h < 16; ++h) {
        int K = (h < 8) ? (kbaseB + h) : (kbaseB + 8 + h);
        float v = 0.0f;
        if (K < 27) {
            int c = K / 9, rem = K % 9;
            int kh = rem / 3, kw = rem % 3;
            int yy = py + kh - 1, xx = px + kw - 1;
            if (yy >= 0 && yy < 32 && xx >= 0 && xx < 32)
                v = x[(c * 32 + yy) * 32 + xx];
        }
        b[h] = (_Float16)v;
    }

    v8f acc = {};
    acc = __builtin_amdgcn_wmma_f32_16x16x32_f16(
        false, a, false, b, (short)0, acc, false, false);

    // D scatter + pixel_shuffle(3) into padded residual
#pragma unroll
    for (int r = 0; r < 8; ++r) {
        int M = mtile * 16 + r + ((lane < 16) ? 0 : 8);
        if (M < 27) {
            int c = M / 9, rem = M % 9, r1 = rem / 3, r2 = rem % 3;
            int Y = py * 3 + r1, X = px * 3 + r2;
            residual[(c * PW + (Y + 1)) * PW + (X + 1)] = acc[r];
        }
    }
}

// ---------------------------------------------------------------------------
// Kernel 2: AdderNet conv3x3 (out = -sum |x - w|), TDM-staged input tile.
//   Block = 256 threads = 32x8 pixel tile, OBLK output channels.
//   2 VALU/elem (v_sub + v_sub|abs|), SGPR weights, weight prefetch.
// ---------------------------------------------------------------------------
template <int OBLK, bool RELU, bool ADD_RES, bool PAD_OUT>
__global__ __launch_bounds__(256)
void adder_conv3x3(const float* __restrict__ in,   // padded [C,98,98]
                   const float* __restrict__ w,    // [O,C,3,3]
                   const float* __restrict__ res,  // padded [O,98,98] or null
                   float* __restrict__ out,        // padded or [O,96,96]
                   int C)
{
    extern __shared__ float tile[];  // C * 340 floats

    const int tid   = threadIdx.x;
    const int tx    = blockIdx.x % NTX;
    const int ty    = (blockIdx.x / NTX) % NTY;
    const int og    = blockIdx.x / (NTX * NTY);
    const int x0    = tx * TW, y0 = ty * TH;
    const int obase = og * OBLK;

    // One TDM op stages the whole [C][10][34] halo tile (wave 0 only; TDM
    // ignores EXEC, so gate with a wave-uniform branch).
    if (tid < 32) {
        tdm_tile_load(in + (y0 * PW + x0), (unsigned)(uintptr_t)tile, C);
        __builtin_amdgcn_s_wait_tensorcnt(0);
    }
    __syncthreads();

    const int lx = tid % TW;
    const int ly = tid / TW;

    float acc[OBLK];
#pragma unroll
    for (int i = 0; i < OBLK; ++i) acc[i] = 0.0f;

    for (int c = 0; c < C; ++c) {
        const float* tch = tile + c * TPC;
        if (c + 1 < C)   // global_prefetch of next channel's weights
            __builtin_prefetch(w + ((size_t)obase * C + c + 1) * 9, 0, 1);
#pragma unroll
        for (int kh = 0; kh < 3; ++kh) {
#pragma unroll
            for (int kw = 0; kw < 3; ++kw) {
                float xv = tch[(ly + kh) * (TW + 2) + (lx + kw)];
                int   k  = kh * 3 + kw;
#pragma unroll
                for (int i = 0; i < OBLK; ++i) {
                    float wv = w[((size_t)(obase + i) * C + c) * 9 + k]; // SGPR
                    acc[i] -= fabsf(xv - wv);   // v_sub ; v_sub with |src|
                }
            }
        }
    }

#pragma unroll
    for (int i = 0; i < OBLK; ++i) {
        float v = acc[i];
        if (RELU) v = fmaxf(v, 0.0f);
        int o = obase + i;
        if (PAD_OUT) {
            out[(o * PW + (y0 + ly + 1)) * PW + (x0 + lx + 1)] = v;
        } else {
            if (ADD_RES) v += res[(o * PW + (y0 + ly + 1)) * PW + (x0 + lx + 1)];
            out[(o * IMG + (y0 + ly)) * IMG + (x0 + lx)] = v;
        }
    }
}

// ---------------------------------------------------------------------------
extern "C" void kernel_launch(void* const* d_in, const int* in_sizes, int n_in,
                              void* d_out, int out_size, void* d_ws, size_t ws_size,
                              hipStream_t stream)
{
    (void)in_sizes; (void)n_in; (void)out_size; (void)ws_size;

    const float* x     = (const float*)d_in[0];  // [1,3,32,32]
    const float* w_up  = (const float*)d_in[1];  // [27,3,3,3]
    const float* w_in  = (const float*)d_in[2];  // [56,3,3,3]
    const float* w_res = (const float*)d_in[3];  // [16,56,56,3,3]
    const float* w_out = (const float*)d_in[4];  // [3,56,3,3]
    float*       outp  = (float*)d_out;          // [1,3,96,96]

    float* ws       = (float*)d_ws;
    float* residual = ws;                        // padded  3*9604
    float* hA       = residual + 3 * PP;         // padded 56*9604
    float* hB       = hA + 56 * PP;              // padded 56*9604

    const int tiles  = NTX * NTY;                // 36
    const size_t lds3  = (size_t)3  * TPC * sizeof(float);
    const size_t lds56 = (size_t)56 * TPC * sizeof(float);   // ~76 KB

    // 0) zero padded buffers (pads must be 0; interiors rewritten below)
    const int nz = (3 + 56 + 56) * PP;
    zero_buf<<<(nz + 255) / 256, 256, 0, stream>>>(ws, nz);

    // 1) upsample conv (WMMA) + pixel shuffle -> padded residual
    upconv_ps_wmma<<<128, 32, 0, stream>>>(x, w_up, residual);

    // 2) h = relu(adder(residual, w_in)) : C=3 -> O=56
    adder_conv3x3<8, true, false, true><<<tiles * 7, 256, lds3, stream>>>(
        residual, w_in, nullptr, hA, 3);

    // 3) 16 residual adder layers : C=56 -> O=56
    float* cur = hA;
    float* nxt = hB;
    for (int l = 0; l < 16; ++l) {
        adder_conv3x3<8, true, false, true><<<tiles * 7, 256, lds56, stream>>>(
            cur, w_res + (size_t)l * 56 * 56 * 9, nullptr, nxt, 56);
        float* t = cur; cur = nxt; nxt = t;
    }

    // 4) out = adder(h, w_out) + residual : C=56 -> O=3, no relu
    adder_conv3x3<3, false, true, false><<<tiles, 256, lds56, stream>>>(
        cur, w_out, residual, outp, 56);
}
